// Attention_87471303950433
// MI455X (gfx1250) — matrix operand order repaired and checked by reading, not compiled, not run on hardware
//
#include <hip/hip_runtime.h>

// ---------------------------------------------------------------------------
// Types / helpers
// ---------------------------------------------------------------------------
typedef __attribute__((ext_vector_type(16))) __bf16 v16bf;
typedef __attribute__((ext_vector_type(8)))  float  v8f;
typedef __attribute__((ext_vector_type(4)))  unsigned int u32x4;
typedef __attribute__((ext_vector_type(8)))  int i32x8;
typedef __attribute__((ext_vector_type(4)))  int i32x4;

union Frag16 {
    v16bf v;
    u32x4 q[2];
};

__device__ inline unsigned short f2bf(float x) {
    __bf16 h = (__bf16)x;                      // RNE convert
    return __builtin_bit_cast(unsigned short, h);
}

__device__ inline v8f wmma_bf16(v16bf a, v16bf b, v8f c) {
    return __builtin_amdgcn_wmma_f32_16x16x32_bf16(
        false, a, false, b, (short)0, c, false, false);
}

// ---------------------------------------------------------------------------
// Tensor Data Mover (CDNA5): DMA a 2-D bf16 tile (32 elems x 128 rows) from
// global into LDS, with LDS row padding 32->40 ushorts. Descriptor layout per
// CDNA5 ISA.
// ---------------------------------------------------------------------------
#if defined(__has_builtin)
#if __has_builtin(__builtin_amdgcn_tensor_load_to_lds) && \
    __has_builtin(__builtin_amdgcn_s_wait_tensorcnt)
#define USE_TDM 1
#endif
#endif
#ifndef USE_TDM
#define USE_TDM 0
#endif

#if USE_TDM
__device__ inline void tdm_load_tile_a(const unsigned short* gsrc,
                                       unsigned ldsByteOff,
                                       unsigned rowStrideElems) {
    const unsigned long long ga = (unsigned long long)(size_t)gsrc;
    // ---- D# group 0 (128b): count=1 | lds_addr | global_addr[56:0] | type=2
    u32x4 g0;
    g0.x = 1u;                                            // count=1, user mode
    g0.y = ldsByteOff;                                    // LDS byte address
    g0.z = (unsigned)ga;                                  // global_addr[31:0]
    g0.w = ((unsigned)(ga >> 32) & 0x01FFFFFFu) | 0x80000000u; // [56:32]|type=2
    // ---- D# group 1 (256b)
    const unsigned w0 = (1u << 16)    // data_size = 2 bytes
                      | (1u << 20)    // pad_enable
                      | (3u << 22)    // pad_interval: 16 DWORDs (one 64B row)
                      | (3u << 25);   // pad_amount: 4 DWORDs (8 ushorts)
    const unsigned td0 = 1u << 30;    // huge tensor dims: never OOB
    const unsigned td1 = 1u << 30;
    const unsigned w1 = (td0 & 0xFFFFu) << 16;            // tensor_dim0[15:0]
    const unsigned w2 = (td0 >> 16) | ((td1 & 0xFFFFu) << 16);
    const unsigned w3 = (td1 >> 16) | (32u << 16);        // tile_dim0 = 32
    const unsigned w4 = 128u;                             // tile_dim1=128, dim2=0
    const unsigned w5 = rowStrideElems;                   // dim0_stride[31:0]
    i32x8 g1 = { (int)w0, (int)w1, (int)w2, (int)w3,
                 (int)w4, (int)w5, 0, 0 };
    i32x4 g2 = {0, 0, 0, 0};
    i32x4 g3 = {0, 0, 0, 0};
#if defined(__clang_major__) && (__clang_major__ >= 23)
    i32x8 g4 = {0, 0, 0, 0, 0, 0, 0, 0};
    __builtin_amdgcn_tensor_load_to_lds(g0, g1, g2, g3, g4, 0);
#else
    __builtin_amdgcn_tensor_load_to_lds(g0, g1, g2, g3, 0);
#endif
}
#endif  // USE_TDM

// ---------------------------------------------------------------------------
// f32 -> bf16 conversion (vectorized x4)
// ---------------------------------------------------------------------------
__global__ __launch_bounds__(256) void cvt_f32_to_bf16(
    const float* __restrict__ src, unsigned short* __restrict__ dst, int n4) {
    int i = blockIdx.x * blockDim.x + threadIdx.x;
    if (i < n4) {
        float4 f = ((const float4*)src)[i];
        ushort4 h;
        h.x = f2bf(f.x); h.y = f2bf(f.y); h.z = f2bf(f.z); h.w = f2bf(f.w);
        ((ushort4*)dst)[i] = h;
    }
}

// ---------------------------------------------------------------------------
// bf16 GEMM: out[M,N] = A[M,K] * W[K,N] + bias  (f32 WMMA accumulate).
// 128 threads (4 waves, 2x2), workgroup tile 128x128, wave tile 64x64, BK=32.
// Ping-pong LDS double buffering: A tiles DMA'd by the TDM (wave 0 issues,
// s_wait_tensorcnt before the publish barrier); W tiles register-prefetched
// and stored transposed. Both overlap the 16 WMMAs of the current buffer.
// Fragment layouts per CDNA5 ISA 7.12.2:
//   A-frag: lane m=lane&15, half=lane>>4; chunks at k = 8h and 16+8h
//   B-frag: lane n=lane&15; same k chunks as A-frag (k = 8h and 16+8h)  (LDS holds W^T)
//   C/D:    reg r, lane -> (m = r + 8h, n = lane&15)
// ---------------------------------------------------------------------------
template <bool OUT_F32>
__global__ __launch_bounds__(128) void gemm_bf16(
    const unsigned short* __restrict__ A,   // [M,K] bf16 bits
    const unsigned short* __restrict__ W,   // [K,N] bf16 bits
    const float* __restrict__ bias,         // [N]
    float* __restrict__ outF,
    unsigned short* __restrict__ outH,
    int M, int K, int N) {
    __shared__ unsigned short Als[2][128][40];   // [buf][m][k]
    __shared__ unsigned short Bls[2][128][40];   // [buf][n][k]  (W^T)

    const int t    = threadIdx.x;
    const int lane = t & 31;
    const int wave = t >> 5;                  // 0..3
    const int wm   = wave >> 1;               // 0..1
    const int wn   = wave & 1;                // 0..1
    const int m0   = blockIdx.y * 128;
    const int n0   = blockIdx.x * 128;
    const int mL   = lane & 15;
    const int hL   = lane >> 4;
    const int kk   = (t >> 3) << 1;           // 0,2,...,30 (B staging row pair)
    const int nc   = (t & 7) << 4;            // 0,16,...,112

    u32x4 pb0a, pb0b, pb1a, pb1b;             // B prefetch: 2 rows x 16
#if !USE_TDM
    u32x4 pa[4];                              // A prefetch: full row of 32
#endif

    auto gloadB = [&](int k0) {
        const unsigned short* gB = W + (size_t)(k0 + kk) * N + n0 + nc;
        pb0a = *(const u32x4*)gB;
        pb0b = *(const u32x4*)(gB + 8);
        pb1a = *(const u32x4*)(gB + N);
        pb1b = *(const u32x4*)(gB + N + 8);
    };
    auto lstoreB = [&](int buf) {
        const unsigned short* s0a = (const unsigned short*)&pb0a;
        const unsigned short* s0b = (const unsigned short*)&pb0b;
        const unsigned short* s1a = (const unsigned short*)&pb1a;
        const unsigned short* s1b = (const unsigned short*)&pb1b;
#pragma unroll
        for (int i = 0; i < 8; ++i) {
            unsigned int lo = (unsigned int)s0a[i] | ((unsigned int)s1a[i] << 16);
            *(unsigned int*)&Bls[buf][nc + i][kk] = lo;
            unsigned int hi = (unsigned int)s0b[i] | ((unsigned int)s1b[i] << 16);
            *(unsigned int*)&Bls[buf][nc + 8 + i][kk] = hi;
        }
    };
#if !USE_TDM
    auto gloadA = [&](int k0) {
        const u32x4* gA = (const u32x4*)(A + (size_t)(m0 + t) * K + k0);
        pa[0] = gA[0]; pa[1] = gA[1]; pa[2] = gA[2]; pa[3] = gA[3];
    };
    auto lstoreA = [&](int buf) {
        u32x4* sA = (u32x4*)&Als[buf][t][0];
        sA[0] = pa[0]; sA[1] = pa[1]; sA[2] = pa[2]; sA[3] = pa[3];
    };
#endif

    v8f acc[4][4];
#pragma unroll
    for (int i = 0; i < 4; ++i)
#pragma unroll
        for (int j = 0; j < 4; ++j) acc[i][j] = {};

    const int nsteps = K >> 5;

    // ---- prologue: stage k0 = 0 into buffer 0 ----
    gloadB(0);
#if USE_TDM
    if (wave == 0)
        tdm_load_tile_a(A + (size_t)m0 * K,
                        (unsigned)(size_t)&Als[0][0][0], (unsigned)K);
#else
    gloadA(0);
    lstoreA(0);
#endif
    lstoreB(0);
#if USE_TDM
    if (wave == 0) __builtin_amdgcn_s_wait_tensorcnt(0);
#endif

    for (int step = 0; step < nsteps; ++step) {
        __syncthreads();                       // buf[cur] ready for all waves
        const int cur = step & 1;
        if (step + 1 < nsteps) {
            gloadB((step + 1) << 5);
#if USE_TDM
            if (wave == 0)
                tdm_load_tile_a(A + (size_t)m0 * K + ((step + 1) << 5),
                                (unsigned)(size_t)&Als[cur ^ 1][0][0],
                                (unsigned)K);
#else
            gloadA((step + 1) << 5);
#endif
        }

        v16bf af[4], bf[4];
#pragma unroll
        for (int i = 0; i < 4; ++i) {
            const unsigned short* p = &Als[cur][wm * 64 + i * 16 + mL][hL * 8];
            Frag16 f;
            f.q[0] = *(const u32x4*)p;
            f.q[1] = *(const u32x4*)(p + 16);
            af[i] = f.v;
        }
#pragma unroll
        for (int j = 0; j < 4; ++j) {
            const unsigned short* p = &Bls[cur][wn * 64 + j * 16 + mL][hL * 8];
            Frag16 f;
            f.q[0] = *(const u32x4*)p;
            f.q[1] = *(const u32x4*)(p + 16);
            bf[j] = f.v;
        }
#pragma unroll
        for (int i = 0; i < 4; ++i)
#pragma unroll
            for (int j = 0; j < 4; ++j)
                acc[i][j] = wmma_bf16(af[i], bf[j], acc[i][j]);

        if (step + 1 < nsteps) {
            lstoreB(cur ^ 1);                  // fill next buffer
#if !USE_TDM
            lstoreA(cur ^ 1);
#endif
#if USE_TDM
            if (wave == 0) __builtin_amdgcn_s_wait_tensorcnt(0);
#endif
        }
    }

    // ---- epilogue: bias + store ----
#pragma unroll
    for (int j = 0; j < 4; ++j) {
        const int n = n0 + wn * 64 + j * 16 + mL;
        const float bv = bias[n];
#pragma unroll
        for (int i = 0; i < 4; ++i) {
#pragma unroll
            for (int r = 0; r < 8; ++r) {
                const int m = m0 + wm * 64 + i * 16 + r + 8 * hL;
                const float val = acc[i][j][r] + bv;
                if (OUT_F32) outF[(size_t)m * N + n] = val;
                else         outH[(size_t)m * N + n] = f2bf(val);
            }
        }
    }
}

// ---------------------------------------------------------------------------
// Flash attention: B=2, N=2048, H=8, DH=128. bf16 Q/K/V ([B*N, C], head-
// interleaved), bf16 context out. Grid (N/128, H, B), 256 threads (8 waves),
// wave owns 16 query rows; 64-key blocks staged in LDS with register
// prefetch of the next block overlapping the current block's WMMAs.
// ---------------------------------------------------------------------------
__global__ __launch_bounds__(256) void attn_kernel(
    const unsigned short* __restrict__ Q,
    const unsigned short* __restrict__ Kt,
    const unsigned short* __restrict__ V,
    unsigned short* __restrict__ Ctx) {
    __shared__ unsigned short Kls[64][136];      // [key][d]
    __shared__ unsigned short Vt[128][72];       // V^T: [d][key]
    __shared__ unsigned short Pls[8][16][72];    // per-wave P tile [m][key]

    const int t    = threadIdx.x;
    const int lane = t & 31;
    const int wave = t >> 5;
    const int mL   = lane & 15;
    const int hL   = lane >> 4;
    const int qblk = blockIdx.x;
    const int h    = blockIdx.y;
    const int b    = blockIdx.z;
    const size_t rowBase = (size_t)b * 2048;
    const int col0 = h * 128;
    const int qrow = qblk * 128 + wave * 16;

    // staging maps
    const int keyK = t >> 2;                     // 0..63
    const int segK = (t & 3) * 32;               // 0,32,64,96
    const int keyV = (t >> 3) * 2;               // 0,2,...,62
    const int cg   = (t & 7) * 16;               // 0,16,...,112

    u32x4 kreg[4], vreg[4];
    auto loadKV = [&](int kb) {
        const u32x4* g =
            (const u32x4*)(Kt + (rowBase + kb + keyK) * 1024 + col0 + segK);
        kreg[0] = g[0]; kreg[1] = g[1]; kreg[2] = g[2]; kreg[3] = g[3];
        const unsigned short* gv =
            V + (rowBase + kb + keyV) * 1024 + col0 + cg;
        vreg[0] = *(const u32x4*)gv;
        vreg[1] = *(const u32x4*)(gv + 8);
        vreg[2] = *(const u32x4*)(gv + 1024);
        vreg[3] = *(const u32x4*)(gv + 1024 + 8);
    };
    auto storeKV = [&]() {
        u32x4* s = (u32x4*)&Kls[keyK][segK];
        s[0] = kreg[0]; s[1] = kreg[1]; s[2] = kreg[2]; s[3] = kreg[3];
        const unsigned short* s0a = (const unsigned short*)&vreg[0];
        const unsigned short* s0b = (const unsigned short*)&vreg[1];
        const unsigned short* s1a = (const unsigned short*)&vreg[2];
        const unsigned short* s1b = (const unsigned short*)&vreg[3];
#pragma unroll
        for (int i = 0; i < 8; ++i) {
            unsigned int lo = (unsigned int)s0a[i] | ((unsigned int)s1a[i] << 16);
            *(unsigned int*)&Vt[cg + i][keyV] = lo;
            unsigned int hi = (unsigned int)s0b[i] | ((unsigned int)s1b[i] << 16);
            *(unsigned int*)&Vt[cg + 8 + i][keyV] = hi;
        }
    };

    // Q fragments: 16 rows x 128 d, 4 A-frags held in registers
    v16bf qf[4];
#pragma unroll
    for (int kk2 = 0; kk2 < 4; ++kk2) {
        const unsigned short* p =
            Q + (rowBase + qrow + mL) * 1024 + col0 + kk2 * 32 + hL * 8;
        Frag16 f;
        f.q[0] = *(const u32x4*)p;
        f.q[1] = *(const u32x4*)(p + 16);
        qf[kk2] = f.v;
    }

    v8f O[8];
#pragma unroll
    for (int c = 0; c < 8; ++c) O[c] = {};
    float mrow[8], lrow[8];
#pragma unroll
    for (int r = 0; r < 8; ++r) { mrow[r] = -1e30f; lrow[r] = 0.0f; }

    const float cs = 0.08838834764831845f * 1.4426950408889634f; // scale*log2e

    loadKV(0);
    for (int kb = 0; kb < 2048; kb += 64) {
        __syncthreads();                // LDS free (prev block consumed)
        storeKV();
        __syncthreads();                // LDS ready
        if (kb + 64 < 2048) loadKV(kb + 64);   // overlap next block w/ WMMAs

        // ---- S = Q K^T : 4 tiles of 16x16 over 64 keys ----
        v8f S[4];
#pragma unroll
        for (int jt = 0; jt < 4; ++jt) {
            v8f s = {};
#pragma unroll
            for (int kk2 = 0; kk2 < 4; ++kk2) {
                const unsigned short* p = &Kls[jt * 16 + mL][kk2 * 32 + hL * 8];
                Frag16 f;
                f.q[0] = *(const u32x4*)p;
                f.q[1] = *(const u32x4*)(p + 16);
                s = wmma_bf16(qf[kk2], f.v, s);
            }
            S[jt] = s;
        }

        // ---- online softmax ----
        float alpha[8];
#pragma unroll
        for (int r = 0; r < 8; ++r) {
            float m = S[0][r];
            m = fmaxf(m, S[1][r]);
            m = fmaxf(m, S[2][r]);
            m = fmaxf(m, S[3][r]);
            m = fmaxf(m, __shfl_xor(m, 1, 32));
            m = fmaxf(m, __shfl_xor(m, 2, 32));
            m = fmaxf(m, __shfl_xor(m, 4, 32));
            m = fmaxf(m, __shfl_xor(m, 8, 32));
            const float mnew = fmaxf(mrow[r], m);
            alpha[r] = exp2f((mrow[r] - mnew) * cs);
            mrow[r] = mnew;
            float sum = 0.0f;
#pragma unroll
            for (int jt = 0; jt < 4; ++jt) {
                const float p = exp2f((S[jt][r] - mnew) * cs);
                Pls[wave][r + 8 * hL][jt * 16 + mL] = f2bf(p);
                sum += p;
            }
            sum += __shfl_xor(sum, 1, 32);
            sum += __shfl_xor(sum, 2, 32);
            sum += __shfl_xor(sum, 4, 32);
            sum += __shfl_xor(sum, 8, 32);
            lrow[r] = lrow[r] * alpha[r] + sum;
        }

        // ---- rescale O, then O += P * V ----
#pragma unroll
        for (int c = 0; c < 8; ++c)
#pragma unroll
            for (int r = 0; r < 8; ++r) O[c][r] *= alpha[r];

#pragma unroll
        for (int k2 = 0; k2 < 2; ++k2) {
            const unsigned short* pp = &Pls[wave][mL][k2 * 32 + hL * 8];
            Frag16 fa;
            fa.q[0] = *(const u32x4*)pp;
            fa.q[1] = *(const u32x4*)(pp + 16);
#pragma unroll
            for (int c = 0; c < 8; ++c) {
                const unsigned short* pv = &Vt[c * 16 + mL][k2 * 32 + hL * 8];
                Frag16 fb;
                fb.q[0] = *(const u32x4*)pv;
                fb.q[1] = *(const u32x4*)(pv + 16);
                O[c] = wmma_bf16(fa.v, fb.v, O[c]);
            }
        }
    }

    // ---- normalize + store context (bf16) ----
#pragma unroll
    for (int c = 0; c < 8; ++c) {
#pragma unroll
        for (int r = 0; r < 8; ++r) {
            const float o = O[c][r] / lrow[r];
            const size_t row = rowBase + qrow + r + 8 * hL;
            Ctx[row * 1024 + col0 + c * 16 + mL] = f2bf(o);
        }
    }
}

// ---------------------------------------------------------------------------
// Launcher
// ---------------------------------------------------------------------------
extern "C" void kernel_launch(void* const* d_in, const int* in_sizes, int n_in,
                              void* d_out, int out_size, void* d_ws, size_t ws_size,
                              hipStream_t stream) {
    (void)in_sizes; (void)n_in; (void)out_size; (void)ws_size;
    const float* Xq  = (const float*)d_in[0];
    const float* Xkv = (const float*)d_in[1];
    const float* Wq  = (const float*)d_in[2];
    const float* bq  = (const float*)d_in[3];
    const float* Wk  = (const float*)d_in[4];
    const float* bk  = (const float*)d_in[5];
    const float* Wv  = (const float*)d_in[6];
    const float* bv  = (const float*)d_in[7];
    const float* Wo  = (const float*)d_in[8];
    const float* bo  = (const float*)d_in[9];
    float* out = (float*)d_out;
    unsigned short* ws = (unsigned short*)d_ws;

    const size_t MC = (size_t)4096 * 1024;   // (B*N) * C
    const size_t CC = (size_t)1024 * 1024;   // C * C
    unsigned short* Xq16  = ws;
    unsigned short* Xkv16 = Xq16 + MC;
    unsigned short* Wq16  = Xkv16 + MC;
    unsigned short* Wk16  = Wq16 + CC;
    unsigned short* Wv16  = Wk16 + CC;
    unsigned short* Wo16  = Wv16 + CC;
    unsigned short* Q16   = Wo16 + CC;
    unsigned short* K16   = Q16 + MC;
    unsigned short* V16   = K16 + MC;
    unsigned short* C16   = V16 + MC;

    auto cvt = [&](const float* s, unsigned short* d, size_t n) {
        int n4 = (int)(n / 4);
        cvt_f32_to_bf16<<<dim3((n4 + 255) / 256), dim3(256), 0, stream>>>(s, d, n4);
    };
    cvt(Xq, Xq16, MC);
    cvt(Xkv, Xkv16, MC);
    cvt(Wq, Wq16, CC);
    cvt(Wk, Wk16, CC);
    cvt(Wv, Wv16, CC);
    cvt(Wo, Wo16, CC);

    dim3 ggrid(1024 / 128, 4096 / 128);
    dim3 gblk(128);
    gemm_bf16<false><<<ggrid, gblk, 0, stream>>>(Xq16,  Wq16, bq, nullptr, Q16, 4096, 1024, 1024);
    gemm_bf16<false><<<ggrid, gblk, 0, stream>>>(Xkv16, Wk16, bk, nullptr, K16, 4096, 1024, 1024);
    gemm_bf16<false><<<ggrid, gblk, 0, stream>>>(Xkv16, Wv16, bv, nullptr, V16, 4096, 1024, 1024);

    attn_kernel<<<dim3(16, 8, 2), dim3(256), 0, stream>>>(Q16, K16, V16, C16);

    gemm_bf16<true><<<ggrid, gblk, 0, stream>>>(C16, Wo16, bo, out, nullptr, 4096, 1024, 1024);
}
